// OpenFoldSeedModel_66520453481113
// MI455X (gfx1250) — matrix operand order repaired and checked by reading, not compiled, not run on hardware
//
#include <hip/hip_runtime.h>
#include <cmath>

// ---------------- CDNA5 WMMA types/helpers ----------------
typedef __attribute__((ext_vector_type(16))) _Float16 v16h;
typedef __attribute__((ext_vector_type(8)))  _Float16 v8h;
typedef __attribute__((ext_vector_type(4)))  _Float16 v4h;
typedef __attribute__((ext_vector_type(8)))  float    v8f;

union HF16 { v16h v; v8h h[2]; _Float16 e[16]; };

__device__ __forceinline__ v16h ld_frag(const _Float16* p0, const _Float16* p1) {
  HF16 u;
  u.h[0] = *(const v8h*)p0;
  u.h[1] = *(const v8h*)p1;
  return u.v;
}

__device__ __forceinline__ v8f wmma16x16x32(v16h a, v16h b, v8f c) {
  // D = A(16x32 f16) * B(32x16 f16) + C(16x16 f32)
  return __builtin_amdgcn_wmma_f32_16x16x32_f16(false, a, false, b, (short)0, c,
                                                false, false);
}

__device__ __forceinline__ int imin(int a, int b) { return a < b ? a : b; }

// ---------------- LayerNorm (one block per row) ----------------
__global__ __launch_bounds__(256) void ln_kernel(const float* __restrict__ x,
                                                 const float* __restrict__ g,
                                                 const float* __restrict__ b,
                                                 float* __restrict__ y, int C) {
  __shared__ float red[256];
  long row = blockIdx.x;
  int t = threadIdx.x;
  const float* xr = x + row * (long)C;
  float v = (t < C) ? xr[t] : 0.f;
  red[t] = v;
  __syncthreads();
  for (int s = 128; s > 0; s >>= 1) { if (t < s) red[t] += red[t + s]; __syncthreads(); }
  float mu = red[0] / (float)C;
  __syncthreads();
  float d = (t < C) ? (v - mu) : 0.f;
  red[t] = d * d;
  __syncthreads();
  for (int s = 128; s > 0; s >>= 1) { if (t < s) red[t] += red[t + s]; __syncthreads(); }
  float rs = rsqrtf(red[0] / (float)C + 1e-5f);
  if (t < C) y[row * (long)C + t] = d * rs * g[t] + b[t];
}

// ---------------- Generic strided NT GEMM with WMMA ----------------
// C[m,n] = act( alpha * sum_k A[m*sAm+k*sAk]*B[n*sBn+k*sBk] + bias[n] ) + res
// Requirement: K % 32 == 0 (all call sites: 32/128/256/1024)
struct GemmP {
  const float* A; const float* B; const float* bias; const float* res; float* C;
  long sAm, sAk, sAb;
  long sBn, sBk, sBb;
  long sCm, sCn, sCb;
  int M, N, K, act;   // act: 0 none, 1 exact GELU
  float alpha;
};

#define LDT 40  // padded LDS row stride (halves); multiple of 8 -> 16B aligned frags

__global__ __launch_bounds__(128) void gemm_nt_wmma(GemmP p) {
  __shared__ __align__(16) _Float16 As[64 * LDT];
  __shared__ __align__(16) _Float16 Bs[64 * LDT];
  const int tid = threadIdx.x;
  const int wid = tid >> 5, lid = tid & 31;
  const int bm = blockIdx.y * 64, bn = blockIdx.x * 64;
  const long ab = (long)blockIdx.z * p.sAb;
  const long bb = (long)blockIdx.z * p.sBb;
  const long cb = (long)blockIdx.z * p.sCb;
  const bool fullM = (bm + 64 <= p.M);
  const bool fullN = (bn + 64 <= p.N);
  const bool avec  = fullM && (p.sAk == 1);   // contiguous-K activations
  const bool bvec1 = fullN && (p.sBn == 1);   // weights: contiguous along N
  const bool bvec2 = fullN && (p.sBk == 1);   // contiguous-K B operand
  // fast-path per-thread coords
  const int am  = tid >> 3;            // 0..15 (+16*rr) : rows
  const int ak  = (tid & 7) << 2;      // 0..28          : k (float4)
  const int bk8 = tid >> 4;            // 0..7 (+8*rr)   : k rows
  const int bn4 = (tid & 15) << 2;     // 0..60          : n (float4)
  // scalar-path per-thread coords: element i -> row (tid>>5)+4i, k = tid&31
  const int sm0 = tid >> 5, skk = tid & 31;
  const float* aScal = p.A + ab + (long)(bm + sm0) * p.sAm + (long)skk * p.sAk;
  const float* bScal = p.B + bb + (long)(bn + sm0) * p.sBn + (long)skk * p.sBk;

  v8f acc[4] = {v8f{}, v8f{}, v8f{}, v8f{}};
  for (int k0 = 0; k0 < p.K; k0 += 32) {
    // ---- stage A tile (64 x 32) as f16, layout [m][k] ----
    if (avec) {
#pragma unroll
      for (int rr = 0; rr < 4; ++rr) {
        int m = am + rr * 16;
        const float4 f = *(const float4*)&p.A[ab + (long)(bm + m) * p.sAm + (k0 + ak)];
        v4h h = {(_Float16)f.x, (_Float16)f.y, (_Float16)f.z, (_Float16)f.w};
        *(v4h*)&As[m * LDT + ak] = h;
      }
      if (k0 + 32 < p.K)
        __builtin_prefetch(&p.A[ab + (long)(bm + am) * p.sAm + (k0 + 32 + ak)], 0, 1);
    } else if (fullM) {
      // uniform-stride pointer walk: no clamps, no selects, adds only
      float tv[16];
#pragma unroll
      for (int i = 0; i < 16; ++i) tv[i] = aScal[(long)(4 * i) * p.sAm];
#pragma unroll
      for (int i = 0; i < 16; ++i) As[(sm0 + 4 * i) * LDT + skk] = (_Float16)tv[i];
      aScal += 32 * p.sAk;
    } else {
      float tv[16];
#pragma unroll
      for (int i = 0; i < 16; ++i) {
        int gm = bm + sm0 + 4 * i;
        float v = p.A[ab + (long)imin(gm, p.M - 1) * p.sAm + (long)(k0 + skk) * p.sAk];
        tv[i] = (gm < p.M) ? v : 0.f;
      }
#pragma unroll
      for (int i = 0; i < 16; ++i) As[(sm0 + 4 * i) * LDT + skk] = (_Float16)tv[i];
    }
    // ---- stage B tile (64 n x 32 k) as f16, layout [n][k] ----
    if (bvec1) {  // e.g. row-major weights W[k*N + n]
#pragma unroll
      for (int rr = 0; rr < 4; ++rr) {
        int k = bk8 + rr * 8;
        const float4 f = *(const float4*)&p.B[bb + (long)(k0 + k) * p.sBk + (bn + bn4)];
        Bs[(bn4 + 0) * LDT + k] = (_Float16)f.x;
        Bs[(bn4 + 1) * LDT + k] = (_Float16)f.y;
        Bs[(bn4 + 2) * LDT + k] = (_Float16)f.z;
        Bs[(bn4 + 3) * LDT + k] = (_Float16)f.w;
      }
      if (k0 + 32 < p.K)
        __builtin_prefetch(&p.B[bb + (long)(k0 + 32 + bk8) * p.sBk + (bn + bn4)], 0, 1);
    } else if (bvec2) {  // B with contiguous k (NT operand)
#pragma unroll
      for (int rr = 0; rr < 4; ++rr) {
        int n = am + rr * 16;
        const float4 f = *(const float4*)&p.B[bb + (long)(bn + n) * p.sBn + (k0 + ak)];
        v4h h = {(_Float16)f.x, (_Float16)f.y, (_Float16)f.z, (_Float16)f.w};
        *(v4h*)&Bs[n * LDT + ak] = h;
      }
    } else if (fullN) {
      float tv[16];
#pragma unroll
      for (int i = 0; i < 16; ++i) tv[i] = bScal[(long)(4 * i) * p.sBn];
#pragma unroll
      for (int i = 0; i < 16; ++i) Bs[(sm0 + 4 * i) * LDT + skk] = (_Float16)tv[i];
      bScal += 32 * p.sBk;
    } else {
      float tv[16];
#pragma unroll
      for (int i = 0; i < 16; ++i) {
        int gn = bn + sm0 + 4 * i;
        float v = p.B[bb + (long)imin(gn, p.N - 1) * p.sBn + (long)(k0 + skk) * p.sBk];
        tv[i] = (gn < p.N) ? v : 0.f;
      }
#pragma unroll
      for (int i = 0; i < 16; ++i) Bs[(sm0 + 4 * i) * LDT + skk] = (_Float16)tv[i];
    }
    __syncthreads();
    // B fragment: lane -> col n = wid*16 + lid%16; k = (lid/16)*8 .. +8 and +16
    const _Float16* bp = &Bs[(wid * 16 + (lid & 15)) * LDT + ((lid >> 4) << 3)];
    v16h bf = ld_frag(bp, bp + 16);
#pragma unroll
    for (int r = 0; r < 4; ++r) {
      // A fragment: lane -> row m = r*16 + lid%16; k = (lid/16)*8 .. +8 and +16
      const _Float16* apf = &As[(r * 16 + (lid & 15)) * LDT + ((lid >> 4) << 3)];
      v16h af = ld_frag(apf, apf + 16);
      acc[r] = wmma16x16x32(af, bf, acc[r]);
    }
    __syncthreads();
  }
  // ---- epilogue: incremental addressing (2 64-bit mults total) ----
  const int col = bn + wid * 16 + (lid & 15);
  if (col < p.N) {
    const float bv = p.bias ? p.bias[col] : 0.f;
    const int rb = (lid >> 4) << 3;
    long off = cb + (long)(bm + rb) * p.sCm + (long)col * p.sCn;
    const long cs = p.sCm;
#pragma unroll
    for (int r = 0; r < 4; ++r) {
      long o = off + (long)(r * 16) * cs;
      const int rowb = bm + r * 16 + rb;
#pragma unroll
      for (int v = 0; v < 8; ++v) {
        if (rowb + v < p.M) {
          float x = acc[r][v] * p.alpha + bv;
          if (p.act == 1) x = 0.5f * x * (1.f + erff(x * 0.7071067811865475f));
          if (p.res) x += p.res[o];
          p.C[o] = x;
        }
        o += cs;
      }
    }
  }
}

// ---------------- Fused multi-head attention (D = 32) ----------------
// One workgroup per (batch, head). Q/K/V/O share addressing:
//   elem(seq s, dim d) at base(b,h) + s*sSeq + d,  base = b*sBatch + h*32
struct AttnP {
  const float* Q; const float* K; const float* V; float* O;
  const float* bias;        // optional logits bias: bias[q*bq + k*bk + h*bh]
  long bq, bk, bh;
  long sBatch, sSeq;
  int S;                    // sequence length (multiple of 32, <= 256)
  float scale;
};

#define KT_LD 40
#define VT_LD 264
#define PR_LD 264
#define LG_LD 257

__global__ __launch_bounds__(128) void attn_wmma(AttnP p) {
  __shared__ __align__(16) _Float16 Kt[256 * KT_LD];  // [key][d]
  __shared__ __align__(16) _Float16 Vt[32 * VT_LD];   // [d][key]
  __shared__ __align__(16) float    Lg[64 * LG_LD];   // per-wave 16 x S logits
  __shared__ __align__(16) _Float16 Pr[64 * PR_LD];   // per-wave 16 x S probs
  const int b = blockIdx.x, h = blockIdx.y;
  const long base = (long)b * p.sBatch + (long)h * 32;
  const int tid = threadIdx.x;
  const int wid = tid >> 5, lid = tid & 31;
  const int S = p.S;
  for (int i = tid; i < S * 32; i += 128) {
    int key = i >> 5, d = i & 31;
    float kv = p.K[base + (long)key * p.sSeq + d];
    float vv = p.V[base + (long)key * p.sSeq + d];
    Kt[key * KT_LD + d] = (_Float16)kv;
    Vt[d * VT_LD + key] = (_Float16)vv;
  }
  __syncthreads();
  const int r = lid & 15, hh = lid >> 4;
  const int nqt = S >> 4;
  for (int qt = wid; qt < nqt; qt += 4) {
    // Q fragment (from global, f32 -> f16, 4x b128 loads)
    HF16 qf;
    {
      const float* qp = p.Q + base + (long)(qt * 16 + r) * p.sSeq + (hh << 3);
      const float4 f0 = *(const float4*)(qp);
      const float4 f1 = *(const float4*)(qp + 4);
      const float4 f2 = *(const float4*)(qp + 16);
      const float4 f3 = *(const float4*)(qp + 20);
      qf.e[0]  = (_Float16)f0.x; qf.e[1]  = (_Float16)f0.y;
      qf.e[2]  = (_Float16)f0.z; qf.e[3]  = (_Float16)f0.w;
      qf.e[4]  = (_Float16)f1.x; qf.e[5]  = (_Float16)f1.y;
      qf.e[6]  = (_Float16)f1.z; qf.e[7]  = (_Float16)f1.w;
      qf.e[8]  = (_Float16)f2.x; qf.e[9]  = (_Float16)f2.y;
      qf.e[10] = (_Float16)f2.z; qf.e[11] = (_Float16)f2.w;
      qf.e[12] = (_Float16)f3.x; qf.e[13] = (_Float16)f3.y;
      qf.e[14] = (_Float16)f3.z; qf.e[15] = (_Float16)f3.w;
    }
    // S = Q K^T, one WMMA per 16x16 tile (K-dim = 32)
    for (int kt = 0; kt < (S >> 4); ++kt) {
      const _Float16* bp = &Kt[(kt * 16 + r) * KT_LD + (hh << 3)];
      v16h bf = ld_frag(bp, bp + 16);
      v8f acc = {};
      acc = wmma16x16x32(qf.v, bf, acc);
#pragma unroll
      for (int v = 0; v < 8; ++v)
        Lg[(wid * 16 + v + (hh << 3)) * LG_LD + kt * 16 + r] = acc[v];
    }
    // softmax: 2 lanes per row, combine with shfl_xor(16) (wave32)
    const int span = S >> 1;
    const int kbeg = hh * span, kend = kbeg + span;
    float* lrow = &Lg[(wid * 16 + r) * LG_LD];
    const int q = qt * 16 + r;
    float mx = -3.0e38f;
    if (p.bias) {
      const float* brow = p.bias + (long)q * p.bq + (long)h * p.bh;
      for (int k = kbeg; k < kend; ++k) {
        float v = lrow[k] * p.scale + brow[(long)k * p.bk];
        lrow[k] = v;
        mx = fmaxf(mx, v);
      }
    } else {
      for (int k = kbeg; k < kend; ++k) {
        float v = lrow[k] * p.scale;
        lrow[k] = v;
        mx = fmaxf(mx, v);
      }
    }
    mx = fmaxf(mx, __shfl_xor(mx, 16, 32));
    float sm = 0.f;
    for (int k = kbeg; k < kend; ++k) {
      float e = __expf(lrow[k] - mx);
      lrow[k] = e;
      sm += e;
    }
    sm += __shfl_xor(sm, 16, 32);
    float inv = 1.f / sm;
    _Float16* prow = &Pr[(wid * 16 + r) * PR_LD];
    for (int k = kbeg; k < kend; ++k) prow[k] = (_Float16)(lrow[k] * inv);
    // O = P V  (two 16-wide d tiles)
    v8f oa0 = {}, oa1 = {};
    for (int kt = 0; kt < (S >> 5); ++kt) {
      const _Float16* ap = &Pr[(wid * 16 + r) * PR_LD + kt * 32 + (hh << 3)];
      v16h af = ld_frag(ap, ap + 16);
      const _Float16* b0 = &Vt[r * VT_LD + kt * 32 + (hh << 3)];
      oa0 = wmma16x16x32(af, ld_frag(b0, b0 + 16), oa0);
      const _Float16* b1 = &Vt[(16 + r) * VT_LD + kt * 32 + (hh << 3)];
      oa1 = wmma16x16x32(af, ld_frag(b1, b1 + 16), oa1);
    }
#pragma unroll
    for (int v = 0; v < 8; ++v) {
      int row = qt * 16 + v + (hh << 3);
      float* op = p.O + base + (long)row * p.sSeq;
      op[r] = oa0[v];
      op[16 + r] = oa1[v];
    }
  }
}

// ---------------- OPM helpers ----------------
__global__ void mean_over_n_kernel(const float* __restrict__ in, float* __restrict__ out,
                                   int N, int LC) {
  int idx = blockIdx.x * blockDim.x + threadIdx.x;
  if (idx >= LC) return;
  float s = 0.f;
  for (int n = 0; n < N; ++n) s += in[(long)n * LC + idx];
  out[idx] = s / (float)N;
}

__global__ void outer_kernel(const float* __restrict__ lft, const float* __restrict__ rgt,
                             float* __restrict__ out, int L, int C) {
  long idx = (long)blockIdx.x * blockDim.x + threadIdx.x;
  long total = (long)L * L * C;
  if (idx >= total) return;
  int c = (int)(idx % C);
  long ij = idx / C;
  int j = (int)(ij % L);
  int i = (int)(ij / L);
  out[idx] = lft[(long)i * C + c] * rgt[(long)j * C + c];
}

// ---------------- Host-side helpers ----------------
static void run_ln(hipStream_t st, const float* x, const float* g, const float* b,
                   float* y, long rows, int C) {
  ln_kernel<<<dim3((unsigned)rows), dim3(256), 0, st>>>(x, g, b, y, C);
}

static void run_gemm(hipStream_t st, const GemmP& p, int batch) {
  dim3 grid((unsigned)((p.N + 63) / 64), (unsigned)((p.M + 63) / 64), (unsigned)batch);
  gemm_nt_wmma<<<grid, dim3(128), 0, st>>>(p);
}

// standard row-major  C(MxN) = A(MxK) @ W(KxN) [+bias][act][+res]
static void gemm_std(hipStream_t st, const float* A, const float* W, const float* bias,
                     const float* res, float* C, int M, int N, int K, int act = 0,
                     float alpha = 1.f) {
  GemmP p{};
  p.A = A;  p.sAm = K; p.sAk = 1; p.sAb = 0;
  p.B = W;  p.sBn = 1; p.sBk = N; p.sBb = 0;
  p.bias = bias; p.res = res; p.C = C;
  p.sCm = N; p.sCn = 1; p.sCb = 0;
  p.M = M; p.N = N; p.K = K; p.act = act; p.alpha = alpha;
  run_gemm(st, p, 1);
}

extern "C" void kernel_launch(void* const* d_in, const int* in_sizes, int n_in,
                              void* d_out, int out_size, void* d_ws, size_t ws_size,
                              hipStream_t stream) {
  (void)in_sizes; (void)n_in; (void)out_size; (void)ws_size;
  const int NSEQ = 128, L = 256, CM = 256, CZ = 128, CTRI = 128;
  const long M_ELEMS = (long)NSEQ * L * CM;   // 8388608
  const long Z_ELEMS = (long)L * L * CZ;      // 8388608
  const float SCALE32 = 0.17677669529663687f; // 1/sqrt(32)

  float* mOut = (float*)d_out;
  float* zOut = mOut + M_ELEMS;
  hipMemcpyAsync(mOut, d_in[0], M_ELEMS * sizeof(float), hipMemcpyDeviceToDevice, stream);
  hipMemcpyAsync(zOut, d_in[1], Z_ELEMS * sizeof(float), hipMemcpyDeviceToDevice, stream);

  // workspace slots (6 x 33.5MB)
  const long SLOT = 8L * 1024 * 1024;
  float* Xb = (float*)d_ws;
  float* Qb = Xb + SLOT;
  float* Kb = Qb + SLOT;
  float* Vb = Kb + SLOT;
  float* Hb = Vb + SLOT;
  float* Ub = Hb + SLOT;

#define F(i) ((const float*)d_in[i])
  // param indices (depth-first dict insertion order after m, z)
  // row: 2..12 | col: 13..22 | mt: 23..28 | opm: 29..36 | tri_out: 37..44
  // tri_in: 45..52 | par: 53..62 | pac: 63..72 | pt: 73..78

  // ---------------- MSA row attention (with pair bias) ----------------
  run_ln(stream, mOut, F(2), F(3), Xb, (long)NSEQ * L, CM);
  gemm_std(stream, Xb, F(4), F(5), nullptr, Qb, NSEQ * L, CM, CM);
  gemm_std(stream, Xb, F(6), F(7), nullptr, Kb, NSEQ * L, CM, CM);
  gemm_std(stream, Xb, F(8), F(9), nullptr, Vb, NSEQ * L, CM, CM);
  gemm_std(stream, zOut, F(12), nullptr, nullptr, Hb, L * L, 8, CZ);  // pb = z @ Wpb
  {
    AttnP a{};
    a.Q = Qb; a.K = Kb; a.V = Vb; a.O = Xb;
    a.bias = Hb; a.bq = (long)L * 8; a.bk = 8; a.bh = 1;
    a.sBatch = (long)L * CM; a.sSeq = CM; a.S = L; a.scale = SCALE32;
    attn_wmma<<<dim3(NSEQ, 8), dim3(128), 0, stream>>>(a);
  }
  gemm_std(stream, Xb, F(10), F(11), mOut, mOut, NSEQ * L, CM, CM);

  // ---------------- MSA column attention ----------------
  run_ln(stream, mOut, F(13), F(14), Xb, (long)NSEQ * L, CM);
  gemm_std(stream, Xb, F(15), F(16), nullptr, Qb, NSEQ * L, CM, CM);
  gemm_std(stream, Xb, F(17), F(18), nullptr, Kb, NSEQ * L, CM, CM);
  gemm_std(stream, Xb, F(19), F(20), nullptr, Vb, NSEQ * L, CM, CM);
  {
    AttnP a{};
    a.Q = Qb; a.K = Kb; a.V = Vb; a.O = Xb;
    a.bias = nullptr; a.bq = 0; a.bk = 0; a.bh = 0;
    a.sBatch = CM; a.sSeq = (long)L * CM; a.S = NSEQ; a.scale = SCALE32;
    attn_wmma<<<dim3(L, 8), dim3(128), 0, stream>>>(a);
  }
  gemm_std(stream, Xb, F(21), F(22), mOut, mOut, NSEQ * L, CM, CM);

  // ---------------- MSA transition (chunked, fused GELU) ----------------
  run_ln(stream, mOut, F(23), F(24), Xb, (long)NSEQ * L, CM);
  for (int c = 0; c < 4; ++c) {
    const float* Ac = Xb + (long)c * 8192 * CM;
    float* Cc = mOut + (long)c * 8192 * CM;
    gemm_std(stream, Ac, F(25), F(26), nullptr, Hb, 8192, 4 * CM, CM, /*gelu*/ 1);
    gemm_std(stream, Hb, F(27), F(28), Cc, Cc, 8192, CM, 4 * CM);
  }

  // ---------------- Outer product mean ----------------
  run_ln(stream, mOut, F(29), F(30), Xb, (long)NSEQ * L, CM);
  gemm_std(stream, Xb, F(31), F(32), nullptr, Qb, NSEQ * L, 32, CM);
  gemm_std(stream, Xb, F(33), F(34), nullptr, Kb, NSEQ * L, 32, CM);
  {
    int LC = L * 32;
    mean_over_n_kernel<<<dim3((LC + 255) / 256), dim3(256), 0, stream>>>(Qb, Ub, NSEQ, LC);
    mean_over_n_kernel<<<dim3((LC + 255) / 256), dim3(256), 0, stream>>>(Kb, Ub + LC, NSEQ, LC);
    long tot = (long)L * L * 32;
    outer_kernel<<<dim3((unsigned)((tot + 255) / 256)), dim3(256), 0, stream>>>(Ub, Ub + LC, Hb, L, 32);
  }
  gemm_std(stream, Hb, F(35), F(36), zOut, zOut, L * L, CZ, 32);

  // ---------------- Triangle multiplications (batched per channel) ----------------
  const float INV_SQRT_L = 0.0625f; // 1/sqrt(256)
  // tri_out:  u[i,j,c] = sum_k a[i,k,c] * b[j,k,c]
  run_ln(stream, zOut, F(37), F(38), Xb, (long)L * L, CZ);
  gemm_std(stream, Xb, F(39), F(40), nullptr, Qb, L * L, CTRI, CZ);
  gemm_std(stream, Xb, F(41), F(42), nullptr, Kb, L * L, CTRI, CZ);
  {
    GemmP p{};
    p.A = Qb; p.sAm = (long)L * CTRI; p.sAk = CTRI; p.sAb = 1;
    p.B = Kb; p.sBn = (long)L * CTRI; p.sBk = CTRI; p.sBb = 1;
    p.bias = nullptr; p.res = nullptr; p.C = Ub;
    p.sCm = (long)L * CTRI; p.sCn = CTRI; p.sCb = 1;
    p.M = L; p.N = L; p.K = L; p.act = 0; p.alpha = INV_SQRT_L;
    run_gemm(stream, p, CTRI);
  }
  gemm_std(stream, Ub, F(43), F(44), zOut, zOut, L * L, CZ, CTRI);

  // tri_in:  u[i,j,c] = sum_k a[k,i,c] * b[k,j,c]
  run_ln(stream, zOut, F(45), F(46), Xb, (long)L * L, CZ);
  gemm_std(stream, Xb, F(47), F(48), nullptr, Qb, L * L, CTRI, CZ);
  gemm_std(stream, Xb, F(49), F(50), nullptr, Kb, L * L, CTRI, CZ);
  {
    GemmP p{};
    p.A = Qb; p.sAm = CTRI; p.sAk = (long)L * CTRI; p.sAb = 1;
    p.B = Kb; p.sBn = CTRI; p.sBk = (long)L * CTRI; p.sBb = 1;
    p.bias = nullptr; p.res = nullptr; p.C = Ub;
    p.sCm = (long)L * CTRI; p.sCn = CTRI; p.sCb = 1;
    p.M = L; p.N = L; p.K = L; p.act = 0; p.alpha = INV_SQRT_L;
    run_gemm(stream, p, CTRI);
  }
  gemm_std(stream, Ub, F(51), F(52), zOut, zOut, L * L, CZ, CTRI);

  // ---------------- Pair attention, rows ----------------
  run_ln(stream, zOut, F(53), F(54), Xb, (long)L * L, CZ);
  gemm_std(stream, Xb, F(55), F(56), nullptr, Qb, L * L, CZ, CZ);
  gemm_std(stream, Xb, F(57), F(58), nullptr, Kb, L * L, CZ, CZ);
  gemm_std(stream, Xb, F(59), F(60), nullptr, Vb, L * L, CZ, CZ);
  {
    AttnP a{};
    a.Q = Qb; a.K = Kb; a.V = Vb; a.O = Xb;
    a.bias = nullptr; a.bq = 0; a.bk = 0; a.bh = 0;
    a.sBatch = (long)L * CZ; a.sSeq = CZ; a.S = L; a.scale = SCALE32;
    attn_wmma<<<dim3(L, 4), dim3(128), 0, stream>>>(a);
  }
  gemm_std(stream, Xb, F(61), F(62), zOut, zOut, L * L, CZ, CZ);

  // ---------------- Pair attention, columns (swapped strides) ----------------
  run_ln(stream, zOut, F(63), F(64), Xb, (long)L * L, CZ);
  gemm_std(stream, Xb, F(65), F(66), nullptr, Qb, L * L, CZ, CZ);
  gemm_std(stream, Xb, F(67), F(68), nullptr, Kb, L * L, CZ, CZ);
  gemm_std(stream, Xb, F(69), F(70), nullptr, Vb, L * L, CZ, CZ);
  {
    AttnP a{};
    a.Q = Qb; a.K = Kb; a.V = Vb; a.O = Xb;
    a.bias = nullptr; a.bq = 0; a.bk = 0; a.bh = 0;
    a.sBatch = CZ; a.sSeq = (long)L * CZ; a.S = L; a.scale = SCALE32;
    attn_wmma<<<dim3(L, 4), dim3(128), 0, stream>>>(a);
  }
  gemm_std(stream, Xb, F(71), F(72), zOut, zOut, L * L, CZ, CZ);

  // ---------------- Pair transition (chunked, fused GELU) ----------------
  run_ln(stream, zOut, F(73), F(74), Xb, (long)L * L, CZ);
  for (int c = 0; c < 4; ++c) {
    const float* Ac = Xb + (long)c * 16384 * CZ;
    float* Cc = zOut + (long)c * 16384 * CZ;
    gemm_std(stream, Ac, F(75), F(76), nullptr, Hb, 16384, 2 * CZ, CZ, /*gelu*/ 1);
    gemm_std(stream, Hb, F(77), F(78), Cc, Cc, 16384, CZ, 2 * CZ);
  }
#undef F
}
